// SpatialBranch_46686294507577
// MI455X (gfx1250) — compile-verified, matrix operand order checked by reading, not run on hardware
//
#include <hip/hip_runtime.h>

typedef __attribute__((ext_vector_type(16))) _Float16 v16h;
typedef __attribute__((ext_vector_type(8)))  _Float16 v8h;
typedef __attribute__((ext_vector_type(8)))  float    v8f;

#define BATCH 512
#define HW    169          // 13*13
#define TOK   49           // 7*7 tokens
#define DM    64           // model dim
#define ROWS  (BATCH*TOK)  // 25088 token rows
#define NPIX  (BATCH*HW)   // 86528 conv output positions

enum { A_ROW = 0, A_IM2COL = 1 };
enum { B_ROW = 0, B_TRANS = 1 };

// ---------------------------------------------------------------------------
// Branchless im2col element fetch (row terms b/oh/ow hoisted by caller).
// k = ci*9 + kh*3 + kw over NHWC [B,13,13,C] input, SAME zero padding.
// ---------------------------------------------------------------------------
__device__ __forceinline__ float im2col_val(const float* __restrict__ in,
                                            int b, int oh, int ow,
                                            int k, int K, int C) {
  const int ci = k / 9;
  const int r  = k - ci * 9;
  const int kh = r / 3, kw = r - kh * 3;
  const int ih = oh + kh - 1, iw = ow + kw - 1;
  const bool ok = (k < K) & ((unsigned)ih < 13u) & ((unsigned)iw < 13u);
  const size_t idx = ok ? ((((size_t)b * 13 + ih) * 13 + iw) * C + ci) : 0;
  const float v = in[idx];
  return ok ? v : 0.0f;
}

// ---------------------------------------------------------------------------
// WMMA GEMM: Out[M,N] = epi(A[M,K] @ B[K,N]).
// Block = 8 waves = 8 consecutive M-tiles sharing one N-group of NT*16 cols.
// B panel is staged once per block into dynamic LDS, transposed to K-major
// f16 and zero-padded to KPAD, so the hot loop reads each B fragment as two
// aligned ds_load_b128 with no masks/converts. A is read as branchless
// aligned float2 global loads (call sites guarantee M%128==0 and lda a
// multiple of 32 with lda >= round32(K)).
// Fragment layout per CDNA5 ISA 7.12.2 (16-bit A, mirrored B, 8-VGPR C/D):
// lane covers K halfs [kb, kb+8) and [kb+16, kb+24), kb = (lane>=16)*8.
// Epilogue: optional bias[n], folded BatchNorm(+ReLU), residual add.
// ---------------------------------------------------------------------------
template <int AM, int BM, int NT>
__global__ __launch_bounds__(256)
void gemm_wmma(const float* __restrict__ A, const float* __restrict__ Bw,
               float* __restrict__ Out,
               int M, int N, int K, int lda, int ldb, int ldo, int im2colC,
               const float* __restrict__ bias, const float* __restrict__ resid,
               const float* __restrict__ bn_g, const float* __restrict__ bn_b,
               const float* __restrict__ bn_m, const float* __restrict__ bn_v)
{
  extern __shared__ _Float16 Bs[];          // [NT*16][KPAD], K-major, padded
  const int KPAD = (K + 31) & ~31;
  const int COLS = NT * 16;
  const int nStrips = M >> 7;               // M / 128
  const int strip = blockIdx.x % nStrips;
  const int ng    = blockIdx.x / nStrips;
  const int nbase = ng * COLS;

  // ---- Stage B panel into LDS (coalesced), f16, zero-padded ----
  if (BM == B_ROW) {
    for (int idx = threadIdx.x; idx < KPAD * COLS; idx += 256) {
      const int k = idx / COLS;             // COLS is compile-time: shift
      const int c = idx - k * COLS;
      const int n = nbase + c;
      float v = 0.0f;
      if (k < K && n < N) v = Bw[(size_t)k * ldb + n];
      Bs[c * KPAD + k] = (_Float16)v;
    }
  } else {  // weights stored [N,K] row-major (conv weights [Cout, Cin*9])
    for (int idx = threadIdx.x; idx < KPAD * COLS; idx += 256) {
      const int c = idx / KPAD;
      const int k = idx - c * KPAD;
      const int n = nbase + c;
      float v = 0.0f;
      if (k < K && n < N) v = Bw[(size_t)n * ldb + k];
      Bs[c * KPAD + k] = (_Float16)v;
    }
  }
  __syncthreads();

  const int lane = threadIdx.x & 31;
  const int half = lane >> 4;
  const int rc   = lane & 15;
  const int mt   = strip * 8 + (threadIdx.x >> 5);
  const int row  = mt * 16 + rc;            // always < M
  const int kb   = half * 8;

  // Hoisted im2col row decomposition
  int ib = 0, ioh = 0, iow = 0;
  if (AM == A_IM2COL) {
    ib = row / HW;
    const int p = row - ib * HW;
    ioh = p / 13;
    iow = p - ioh * 13;
  }

  v8f acc[NT];
#pragma unroll
  for (int t = 0; t < NT; ++t) { v8f z = {}; acc[t] = z; }

  for (int k0 = 0; k0 < K; k0 += 32) {
    // ---- A fragment (shared across NT N-tiles) ----
    v16h af;
#pragma unroll
    for (int v = 0; v < 8; ++v) {
      const int kp = k0 + kb + ((v < 4) ? (2 * v) : (16 + 2 * (v - 4)));
      float a0, a1;
      if (AM == A_ROW) {
        const float2 av = *reinterpret_cast<const float2*>(
            A + (size_t)row * lda + kp);    // aligned, inside padded row
        a0 = (kp     < K) ? av.x : 0.0f;
        a1 = (kp + 1 < K) ? av.y : 0.0f;
      } else {
        a0 = im2col_val(A, ib, ioh, iow, kp,     K, im2colC);
        a1 = im2col_val(A, ib, ioh, iow, kp + 1, K, im2colC);
      }
      af[2 * v]     = (_Float16)a0;
      af[2 * v + 1] = (_Float16)a1;
    }
    if (AM == A_ROW && k0 + 32 < K)
      __builtin_prefetch(A + (size_t)row * lda + k0 + 32, 0, 1);

    // ---- B fragments from LDS: two aligned b128 reads + WMMA ----
#pragma unroll
    for (int t = 0; t < NT; ++t) {
      const _Float16* bp = &Bs[(t * 16 + rc) * KPAD + k0 + kb];
      const v8h lo = *reinterpret_cast<const v8h*>(bp);
      const v8h hi = *reinterpret_cast<const v8h*>(bp + 16);
      const v16h bf = __builtin_shufflevector(lo, hi, 0, 1, 2, 3, 4, 5, 6, 7,
                                              8, 9, 10, 11, 12, 13, 14, 15);
      acc[t] = __builtin_amdgcn_wmma_f32_16x16x32_f16(false, af, false, bf,
                                                      (short)0, acc[t],
                                                      false, false);
    }
  }

  // ---- Epilogue ----
#pragma unroll
  for (int t = 0; t < NT; ++t) {
    const int n = nbase + t * 16 + rc;
    if (n < N) {
#pragma unroll
      for (int r = 0; r < 8; ++r) {
        const int m = mt * 16 + r + half * 8;  // always < M
        float v = acc[t][r];
        if (bn_g) {  // fused conv-bias + BatchNorm(eval) + ReLU
          const float inv = rsqrtf(bn_v[n] + 1e-5f);
          v = (v + bias[n] - bn_m[n]) * (bn_g[n] * inv) + bn_b[n];
          v = fmaxf(v, 0.0f);
        } else if (bias) {
          v += bias[n];
        }
        if (resid) v += resid[(size_t)m * ldo + n];
        Out[(size_t)m * ldo + n] = v;
      }
    }
  }
}

// ---------------------------------------------------------------------------
// Elementwise / small kernels
// ---------------------------------------------------------------------------
__global__ void k_nchw_to_nhwc(const float* __restrict__ in, float* __restrict__ out, int C) {
  const int i = blockIdx.x * blockDim.x + threadIdx.x;
  const int total = BATCH * HW * C;
  if (i >= total) return;
  const int b = i / (HW * C);
  const int r = i - b * HW * C;
  const int p = r / C;
  const int c = r - p * C;
  out[i] = in[((size_t)b * C + c) * HW + p];
}

// NHWC [B,13,13,64] -> pooled [B*64, 49] stored with ld 64 (padded for WMMA A)
__global__ void k_pool(const float* __restrict__ in, float* __restrict__ out) {
  const int i = blockIdx.x * blockDim.x + threadIdx.x;
  if (i >= BATCH * DM * TOK) return;
  const int b  = i / (DM * TOK);
  const int r  = i - b * DM * TOK;
  const int c  = r / TOK;
  const int op = r - c * TOK;
  const int oh = op / 7, ow = op - oh * 7;
  const int hs = (oh * 13) / 7, he = ((oh + 1) * 13 + 6) / 7;
  const int wsb = (ow * 13) / 7, we = ((ow + 1) * 13 + 6) / 7;
  float s = 0.0f;
  for (int h = hs; h < he; ++h)
    for (int w = wsb; w < we; ++w)
      s += in[((size_t)b * HW + h * 13 + w) * DM + c];
  out[((size_t)b * DM + c) * 64 + op] = s * (1.0f / (float)((he - hs) * (we - wsb)));
}

// emb [B,64,49] (ld 49) -> tokens [B,49,64] + pos_embed[:, :49]
__global__ void k_tok(const float* __restrict__ e, const float* __restrict__ pos,
                      float* __restrict__ out) {
  const int i = blockIdx.x * blockDim.x + threadIdx.x;
  if (i >= ROWS * DM) return;
  const int b = i / (TOK * DM);
  const int r = i - b * TOK * DM;
  const int l = r / DM;
  const int d = r - l * DM;
  out[i] = e[((size_t)b * DM + d) * TOK + l] + pos[l * DM + d];
}

// LayerNorm over 64 with optional residual input, optional ReLU,
// output at Out[row*ldo + colOff + d].
__global__ void k_ln(const float* __restrict__ x, const float* __restrict__ resid,
                     const float* __restrict__ g, const float* __restrict__ beta,
                     float* __restrict__ out, int rows, int ldo, int colOff, int doRelu) {
  const int rI = blockIdx.x * blockDim.x + threadIdx.x;
  if (rI >= rows) return;
  float buf[DM];
  float mean = 0.0f;
#pragma unroll
  for (int d = 0; d < DM; ++d) {
    float t = x[(size_t)rI * DM + d];
    if (resid) t += resid[(size_t)rI * DM + d];
    buf[d] = t;
    mean += t;
  }
  mean *= (1.0f / DM);
  float var = 0.0f;
#pragma unroll
  for (int d = 0; d < DM; ++d) {
    const float dv = buf[d] - mean;
    var += dv * dv;
  }
  var *= (1.0f / DM);
  const float inv = rsqrtf(var + 1e-5f);
#pragma unroll
  for (int d = 0; d < DM; ++d) {
    float v = (buf[d] - mean) * inv * g[d] + beta[d];
    if (doRelu) v = fmaxf(v, 0.0f);
    out[(size_t)rI * ldo + colOff + d] = v;
  }
}

// Causal depthwise conv (width 4) over L=49 + bias + SiLU.
// xc is columns [0,64) of xz[ROWS,128]; writes y[ROWS,64].
__global__ void k_conv1d(const float* __restrict__ xz, const float* __restrict__ cw,
                         const float* __restrict__ cb, float* __restrict__ y) {
  const int i = blockIdx.x * blockDim.x + threadIdx.x;
  if (i >= ROWS * DM) return;
  const int row = i >> 6;
  const int d   = i & 63;
  const int b   = row / TOK;
  const int l   = row - b * TOK;
  float acc = cb[d];
#pragma unroll
  for (int j = 0; j < 4; ++j) {
    const int ll = l - 3 + j;
    const float xv = xz[((size_t)b * TOK + (ll >= 0 ? ll : 0)) * 128 + d];
    acc += (ll >= 0 ? cw[d * 4 + j] : 0.0f) * xv;
  }
  y[i] = acc / (1.0f + expf(-acc));  // SiLU
}

// dt = softplus(dt_r @ w_dt + b_dt); dt_r = dbc[:, 0:4] (ld 48)
__global__ void k_dt(const float* __restrict__ dbc, const float* __restrict__ wdt,
                     const float* __restrict__ bdt, float* __restrict__ dt) {
  const int i = blockIdx.x * blockDim.x + threadIdx.x;
  if (i >= ROWS * DM) return;
  const int row = i >> 6;
  const int d   = i & 63;
  float s = bdt[d];
#pragma unroll
  for (int r = 0; r < 4; ++r) s += dbc[(size_t)row * 48 + r] * wdt[r * DM + d];
  dt[i] = (s > 20.0f) ? s : log1pf(expf(s));
}

// Selective scan. One block per batch, one thread per channel d.
// B/C token vectors (dbc cols 4..35, ld 48) staged in LDS once.
// Fuses ys + y*dskip and the silu(z) gate; writes gated result over y.
__global__ __launch_bounds__(64)
void k_scan(const float* __restrict__ dbc, const float* __restrict__ dtv,
            const float* __restrict__ xz, const float* __restrict__ a_log,
            const float* __restrict__ dskip, float* __restrict__ y) {
  __shared__ float sB[TOK][16];
  __shared__ float sC[TOK][16];
  const int b = blockIdx.x;
  const int d = threadIdx.x;
  for (int t = d; t < TOK * 32; t += 64) {
    const int l = t >> 5, j = t & 31;
    const float v = dbc[((size_t)b * TOK + l) * 48 + 4 + j];
    if (j < 16) sB[l][j] = v; else sC[l][j - 16] = v;
  }
  __syncthreads();
  float Areg[16];
#pragma unroll
  for (int n = 0; n < 16; ++n) Areg[n] = -expf(a_log[d * 16 + n]);
  float h[16];
#pragma unroll
  for (int n = 0; n < 16; ++n) h[n] = 0.0f;
  const float skip = dskip[d];
  for (int l = 0; l < TOK; ++l) {
    const size_t row = (size_t)b * TOK + l;
    const float dt = dtv[row * DM + d];
    const float yv = y[row * DM + d];
    float ys = 0.0f;
#pragma unroll
    for (int n = 0; n < 16; ++n) {
      h[n] = h[n] * expf(dt * Areg[n]) + dt * sB[l][n] * yv;
      ys = fmaf(h[n], sC[l][n], ys);
    }
    const float z = xz[row * 128 + DM + d];
    const float gate = z / (1.0f + expf(-z));
    y[row * DM + d] = (ys + yv * skip) * gate;
  }
}

// ---------------------------------------------------------------------------
// Host side
// ---------------------------------------------------------------------------
struct MambaP {
  const float *ln_g, *ln_b, *w_in, *conv_w, *conv_b, *w_x,
              *w_dt, *b_dt, *a_log, *dskip, *w_out, *b_out;
};

static MambaP mamba_at(void* const* d, int i) {
  MambaP p;
  p.ln_g  = (const float*)d[i + 0];  p.ln_b  = (const float*)d[i + 1];
  p.w_in  = (const float*)d[i + 2];  p.conv_w= (const float*)d[i + 3];
  p.conv_b= (const float*)d[i + 4];  p.w_x   = (const float*)d[i + 5];
  p.w_dt  = (const float*)d[i + 6];  p.b_dt  = (const float*)d[i + 7];
  p.a_log = (const float*)d[i + 8];  p.dskip = (const float*)d[i + 9];
  p.w_out = (const float*)d[i +10];  p.b_out = (const float*)d[i +11];
  return p;
}

// A row-major GEMM; M%128==0, lda multiple of 32 with lda >= round32(K).
static inline void gemm_row(hipStream_t s, const float* A, const float* B, float* O,
                            int M, int N, int K, int lda, int ldb, int ldo,
                            const float* bias, const float* resid) {
  const int ntN  = (N + 15) / 16;
  const int KPAD = (K + 31) & ~31;
  if ((ntN & 1) == 0) {
    const int blocks = (M >> 7) * (ntN / 2);
    gemm_wmma<A_ROW, B_ROW, 2><<<blocks, 256, 32 * KPAD * 2, s>>>(
        A, B, O, M, N, K, lda, ldb, ldo, 0, bias, resid,
        nullptr, nullptr, nullptr, nullptr);
  } else {
    const int blocks = (M >> 7) * ntN;
    gemm_wmma<A_ROW, B_ROW, 1><<<blocks, 256, 16 * KPAD * 2, s>>>(
        A, B, O, M, N, K, lda, ldb, ldo, 0, bias, resid,
        nullptr, nullptr, nullptr, nullptr);
  }
}

static inline void gemm_conv(hipStream_t s, const float* in, int C, const float* W,
                             float* O, int N, int K, const float* cb, const float* g,
                             const float* beta, const float* m, const float* v) {
  const int M = NPIX;                 // 86528 = 676*128
  const int KPAD = (K + 31) & ~31;
  const int blocks = (M >> 7) * (((N + 15) / 16) / 2);  // N in {32,64}
  gemm_wmma<A_IM2COL, B_TRANS, 2><<<blocks, 256, 32 * KPAD * 2, s>>>(
      in, W, O, M, N, K, 0, K, N, C, cb, nullptr, g, beta, m, v);
}

static void run_mamba(hipStream_t s, float* x_io, const MambaP& p,
                      float* t_h, float* t_xz, float* t_y, float* t_dbc, float* t_dt) {
  k_ln<<<(ROWS + 255) / 256, 256, 0, s>>>(x_io, nullptr, p.ln_g, p.ln_b, t_h, ROWS, DM, 0, 0);
  gemm_row(s, t_h, p.w_in, t_xz, ROWS, 128, 64, 64, 128, 128, nullptr, nullptr);
  k_conv1d<<<(ROWS * DM + 255) / 256, 256, 0, s>>>(t_xz, p.conv_w, p.conv_b, t_y);
  gemm_row(s, t_y, p.w_x, t_dbc, ROWS, 36, 64, 64, 36, 48, nullptr, nullptr);
  k_dt<<<(ROWS * DM + 255) / 256, 256, 0, s>>>(t_dbc, p.w_dt, p.b_dt, t_dt);
  k_scan<<<BATCH, 64, 0, s>>>(t_dbc, t_dt, t_xz, p.a_log, p.dskip, t_y);
  // out = x + (gated @ w_out + b_out), in place over x_io
  gemm_row(s, t_y, p.w_out, x_io, ROWS, 64, 64, 64, 64, 64, p.b_out, x_io);
}

extern "C" void kernel_launch(void* const* d_in, const int* in_sizes, int n_in,
                              void* d_out, int out_size, void* d_ws, size_t ws_size,
                              hipStream_t stream) {
  // Leaf order: depth-first insertion order of setup_inputs() pytree.
  const float* x1    = (const float*)d_in[0];   // [512,30,13,13]
  const float* x2    = (const float*)d_in[1];   // [512,1,13,13] (C=1 => already NHWC)
  const float* c1_w  = (const float*)d_in[2];
  const float* c1_b  = (const float*)d_in[3];
  const float* c1_g  = (const float*)d_in[4];
  const float* c1_bt = (const float*)d_in[5];
  const float* c1_m  = (const float*)d_in[6];
  const float* c1_v  = (const float*)d_in[7];
  const float* c1b_w = (const float*)d_in[8];
  const float* c1b_b = (const float*)d_in[9];
  const float* c1b_g = (const float*)d_in[10];
  const float* c1b_bt= (const float*)d_in[11];
  const float* c1b_m = (const float*)d_in[12];
  const float* c1b_v = (const float*)d_in[13];
  const float* c2_w  = (const float*)d_in[14];
  const float* c2_b  = (const float*)d_in[15];
  const float* c2_g  = (const float*)d_in[16];
  const float* c2_bt = (const float*)d_in[17];
  const float* c2_m  = (const float*)d_in[18];
  const float* c2_v  = (const float*)d_in[19];
  const float* c2b_w = (const float*)d_in[20];
  const float* c2b_b = (const float*)d_in[21];
  const float* c2b_g = (const float*)d_in[22];
  const float* c2b_bt= (const float*)d_in[23];
  const float* c2b_m = (const float*)d_in[24];
  const float* c2b_v = (const float*)d_in[25];
  const float* emb_w1= (const float*)d_in[26];
  const float* emb_b1= (const float*)d_in[27];
  const float* emb_w2= (const float*)d_in[28];
  const float* emb_b2= (const float*)d_in[29];
  const float* pos   = (const float*)d_in[30];  // [1,50,64]
  const float* fm1_l1w = (const float*)d_in[31];
  const float* fm1_l1b = (const float*)d_in[32];
  const MambaP fm1_ms   = mamba_at(d_in, 33);
  const float* fm1_ng = (const float*)d_in[45];
  const float* fm1_nb = (const float*)d_in[46];
  const MambaP fm1_spec = mamba_at(d_in, 47);
  const float* fm2_l1w = (const float*)d_in[59];
  const float* fm2_l1b = (const float*)d_in[60];
  const MambaP fm2_ms   = mamba_at(d_in, 61);
  const float* fm2_ng = (const float*)d_in[73];
  const float* fm2_nb = (const float*)d_in[74];
  const float* fus_w  = (const float*)d_in[75]; // [128,64]
  const float* fus_b  = (const float*)d_in[76];
  float* out = (float*)d_out;                   // [512,49,64]

  // Workspace layout (~124 MB of f32)
  float* ws = (float*)d_ws;
  size_t off = 0;
  auto alloc = [&](size_t n) { float* p = ws + off; off += n; return p; };
  float* t_nhwc = alloc((size_t)NPIX * 30);       // x1 in NHWC
  float* t_c1   = alloc((size_t)NPIX * 32);       // conv stage 1 out (NHWC)
  float* t_c2   = alloc((size_t)NPIX * 64);       // conv stage 2 out (NHWC)
  float* t_pa   = alloc((size_t)BATCH * DM * 64); // pooled, ld 64 (padded)
  float* t_pb   = alloc((size_t)BATCH * DM * 64);
  float* t_e    = alloc((size_t)BATCH * DM * TOK);// emb scratch / LN scratch
  float* t_ta   = alloc((size_t)ROWS * DM);       // tokens branch a
  float* t_tb   = alloc((size_t)ROWS * DM);       // tokens branch b
  float* t_xp   = alloc((size_t)ROWS * DM);       // mamba running state
  float* t_xz   = alloc((size_t)ROWS * 128);
  float* t_y    = alloc((size_t)ROWS * DM);
  float* t_dbc  = alloc((size_t)ROWS * 48);
  float* t_dt   = alloc((size_t)ROWS * DM);
  float* t_cat  = alloc((size_t)ROWS * 128);      // concat of branch outputs
  (void)in_sizes; (void)n_in; (void)out_size; (void)ws_size;

  // ---- Branch A: conv stack -> pool -> emb -> tokens ----
  k_nchw_to_nhwc<<<(NPIX * 30 + 255) / 256, 256, 0, stream>>>(x1, t_nhwc, 30);
  gemm_conv(stream, t_nhwc, 30, c1_w,  t_c1, 32, 270, c1_b,  c1_g,  c1_bt,  c1_m,  c1_v);
  gemm_conv(stream, t_c1,   32, c1b_w, t_c2, 64, 288, c1b_b, c1b_g, c1b_bt, c1b_m, c1b_v);
  k_pool<<<(BATCH * DM * TOK + 255) / 256, 256, 0, stream>>>(t_c2, t_pa);
  gemm_row(stream, t_pa, emb_w1, t_e, BATCH * DM, TOK, TOK, 64, TOK, TOK, emb_b1, nullptr);
  k_tok<<<(ROWS * DM + 255) / 256, 256, 0, stream>>>(t_e, pos, t_ta);

  // ---- Branch B ----
  gemm_conv(stream, x2,   1,  c2_w,  t_c1, 32, 9,   c2_b,  c2_g,  c2_bt,  c2_m,  c2_v);
  gemm_conv(stream, t_c1, 32, c2b_w, t_c2, 64, 288, c2b_b, c2b_g, c2b_bt, c2b_m, c2b_v);
  k_pool<<<(BATCH * DM * TOK + 255) / 256, 256, 0, stream>>>(t_c2, t_pb);
  gemm_row(stream, t_pb, emb_w2, t_e, BATCH * DM, TOK, TOK, 64, TOK, TOK, emb_b2, nullptr);
  k_tok<<<(ROWS * DM + 255) / 256, 256, 0, stream>>>(t_e, pos, t_tb);

  // ---- Fuse branch A (spatial + spectral Mamba) ----
  gemm_row(stream, t_ta, fm1_l1w, t_xp, ROWS, 64, 64, 64, 64, 64, fm1_l1b, nullptr);
  run_mamba(stream, t_xp, fm1_ms,   t_e, t_xz, t_y, t_dbc, t_dt);
  run_mamba(stream, t_xp, fm1_spec, t_e, t_xz, t_y, t_dbc, t_dt);
  k_ln<<<(ROWS + 255) / 256, 256, 0, stream>>>(t_ta, t_xp, fm1_ng, fm1_nb,
                                               t_cat, ROWS, 128, 0, 1);

  // ---- Fuse branch B (spatial Mamba only) ----
  gemm_row(stream, t_tb, fm2_l1w, t_xp, ROWS, 64, 64, 64, 64, 64, fm2_l1b, nullptr);
  run_mamba(stream, t_xp, fm2_ms, t_e, t_xz, t_y, t_dbc, t_dt);
  k_ln<<<(ROWS + 255) / 256, 256, 0, stream>>>(t_tb, t_xp, fm2_ng, fm2_nb,
                                               t_cat, ROWS, 128, 64, 1);

  // ---- Fusion projection: [ROWS,128] @ [128,64] + b -> out ----
  gemm_row(stream, t_cat, fus_w, out, ROWS, 64, 128, 128, 64, 64, fus_b, nullptr);
}
